// SequentialPeAttention_60043642798653
// MI455X (gfx1250) — matrix operand order repaired and checked by reading, not compiled, not run on hardware
//
#include <hip/hip_runtime.h>
#include <hip/hip_bf16.h>

typedef __attribute__((ext_vector_type(16))) _Float16 v16h;
typedef __attribute__((ext_vector_type(4)))  _Float16 v4h;
typedef __attribute__((ext_vector_type(8)))  float    v8f;

#define B_  8
#define S_  1024
#define D_  1024
#define H_  16
#define HD_ 64

__device__ inline v8f zero8() {
  v8f z = {0.f, 0.f, 0.f, 0.f, 0.f, 0.f, 0.f, 0.f};
  return z;
}

// ---------------------------------------------------------------------------
// f32 -> f16 conversion (A-side activations), 4 elements/thread
// ---------------------------------------------------------------------------
__global__ void cvt_f32_to_f16(const float* __restrict__ in,
                               _Float16* __restrict__ out, int n4) {
  int i = blockIdx.x * blockDim.x + threadIdx.x;
  if (i < n4) {
    float4 f = ((const float4*)in)[i];
    v4h h = {(_Float16)f.x, (_Float16)f.y, (_Float16)f.z, (_Float16)f.w};
    *(v4h*)(out + 4 * (size_t)i) = h;
  }
}

// ---------------------------------------------------------------------------
// Weight transpose + convert: in f32 [K,N] row-major -> out f16 [N,K].
// 32x32 tile via padded LDS; 256 threads = 32x8.
// ---------------------------------------------------------------------------
__global__ __launch_bounds__(256) void transpose_cvt_f32_f16(
    const float* __restrict__ in, _Float16* __restrict__ out, int K, int N) {
  __shared__ float tile[32][33];
  const int n0 = blockIdx.x * 32;
  const int k0 = blockIdx.y * 32;
  const int tx = threadIdx.x & 31;
  const int ty = threadIdx.x >> 5;  // 0..7
#pragma unroll
  for (int r = 0; r < 4; ++r)
    tile[ty + 8 * r][tx] = in[(size_t)(k0 + ty + 8 * r) * N + n0 + tx];
  __syncthreads();
#pragma unroll
  for (int r = 0; r < 4; ++r)
    out[(size_t)(n0 + ty + 8 * r) * K + k0 + tx] = (_Float16)tile[tx][ty + 8 * r];
}

// ---------------------------------------------------------------------------
// V transpose: qkv16 v-part [b,s,h,d] -> vT[(b*H+h)*HD + d][s]  (f16)
// grid (S/64, B*H), 256 threads, 64x64 tile.
// ---------------------------------------------------------------------------
__global__ __launch_bounds__(256) void transpose_v_f16(
    const _Float16* __restrict__ qkv, _Float16* __restrict__ vT) {
  __shared__ alignas(16) _Float16 t[64][72];  // padded rows
  const int sb  = blockIdx.x * 64;
  const int bh  = blockIdx.y;
  const int b   = bh / H_;
  const int h   = bh % H_;
  const int tid = threadIdx.x;
#pragma unroll
  for (int rep = 0; rep < 2; ++rep) {
    int c = tid + rep * 256;
    int r = c >> 3, q = (c & 7) * 8;
    *(uint4*)&t[r][q] =
        *(const uint4*)&qkv[(size_t)(b * S_ + sb + r) * (3 * D_) + 2 * D_ + h * HD_ + q];
  }
  __syncthreads();
  const int d  = tid >> 2;
  const int s0 = (tid & 3) * 16;
  size_t base = ((size_t)bh * HD_ + d) * S_ + sb + s0;
#pragma unroll
  for (int g = 0; g < 4; ++g) {
    v4h o = {t[s0 + 4 * g + 0][d], t[s0 + 4 * g + 1][d],
             t[s0 + 4 * g + 2][d], t[s0 + 4 * g + 3][d]};
    *(v4h*)&vT[base + 4 * g] = o;
  }
}

// ---------------------------------------------------------------------------
// WMMA GEMM: C[M,N] = A[M,K] * Bt[N,K]^T + bias   (Bt pre-transposed)
// 256 threads (8 waves). Block tile 256(M) x 64(N), K step 32.
// Each wave: 32x64 strip = 2 A-frags x 4 B-frags = 8 WMMA / K-step
// (1.5 ds_load_b128 per WMMA; every B-frag feeds two independent WMMAs).
// ---------------------------------------------------------------------------
__global__ __launch_bounds__(256) void gemm_f16_wmma(
    const _Float16* __restrict__ A,    // [M,K] f16
    const _Float16* __restrict__ Bt,   // [N,K] f16
    const float*    __restrict__ bias, // [N]   f32
    _Float16* __restrict__ outH,       // f16 out (if outHalf)
    float*    __restrict__ outF,       // f32 out (if !outHalf)
    int M, int N, int K, int outHalf) {
  __shared__ alignas(32) _Float16 sA[256 * 32];   // [m][k]
  __shared__ alignas(32) _Float16 sBt[64 * 32];   // [n][k]

  const int tid  = threadIdx.x;
  const int lane = tid & 31;
  const int wave = tid >> 5;  // 0..7
  const int m0   = blockIdx.y * 256;
  const int n0   = blockIdx.x * 64;
  const int mlo  = lane & 15;
  const int khf  = lane >> 4;

  v8f acc[2][4];
#pragma unroll
  for (int mt = 0; mt < 2; ++mt)
#pragma unroll
    for (int nt = 0; nt < 4; ++nt) acc[mt][nt] = zero8();

  for (int k0 = 0; k0 < K; k0 += 32) {
    __syncthreads();
    // stage A tile 256x32 halves (4 x b128 per thread)
#pragma unroll
    for (int rep = 0; rep < 4; ++rep) {
      int c = tid + rep * 256;
      int r = c >> 2, q = (c & 3) * 8;
      *(uint4*)&sA[r * 32 + q] = *(const uint4*)&A[(size_t)(m0 + r) * K + k0 + q];
    }
    // stage Bt tile 64x32 halves (1 x b128 per thread)
    {
      int r = tid >> 2, q = (tid & 3) * 8;
      *(uint4*)&sBt[r * 32 + q] = *(const uint4*)&Bt[(size_t)(n0 + r) * K + k0 + q];
    }
    // prefetch next K tile into GL2 while this one is consumed
    if (k0 + 32 < K) {
      __builtin_prefetch(&A[(size_t)(m0 + tid) * K + k0 + 32], 0, 0);
      __builtin_prefetch(&Bt[(size_t)(n0 + (tid >> 2)) * K + k0 + 32], 0, 0);
    }
    __syncthreads();

    v16h af0 = *(const v16h*)&sA[(wave * 32 + mlo) * 32 + 16 * khf];
    v16h af1 = *(const v16h*)&sA[(wave * 32 + 16 + mlo) * 32 + 16 * khf];
#pragma unroll
    for (int nt = 0; nt < 4; ++nt) {
      v16h bf = *(const v16h*)&sBt[(nt * 16 + mlo) * 32 + 16 * khf];
      acc[0][nt] = __builtin_amdgcn_wmma_f32_16x16x32_f16(
          false, af0, false, bf, (short)0, acc[0][nt], false, false);
      acc[1][nt] = __builtin_amdgcn_wmma_f32_16x16x32_f16(
          false, af1, false, bf, (short)0, acc[1][nt], false, false);
    }
  }

  // C layout: VGPR i -> row = i + 8*khf, col = mlo
#pragma unroll
  for (int mt = 0; mt < 2; ++mt)
#pragma unroll
    for (int nt = 0; nt < 4; ++nt) {
      int col = n0 + nt * 16 + mlo;
      float bv = bias[col];
#pragma unroll
      for (int i = 0; i < 8; ++i) {
        int row = m0 + wave * 32 + mt * 16 + i + 8 * khf;
        float v = acc[mt][nt][i] + bv;
        if (outHalf)
          outH[(size_t)row * N + col] = (_Float16)v;
        else
          outF[(size_t)row * N + col] = v;
      }
    }
}

// ---------------------------------------------------------------------------
// Flash attention: 256 threads (8 waves) handle 128 q rows of one (b,h);
// each wave owns 16 q rows. K tile [kv][d]; V staged transposed [d][kv];
// all WMMA fragments are contiguous 32B LDS reads.
// ---------------------------------------------------------------------------
__global__ __launch_bounds__(256) void flash_attn_wmma(
    const _Float16* __restrict__ qkv,  // [B*S, 3D] f16
    const _Float16* __restrict__ vT,   // [(B*H)*HD, S] f16
    const float*    __restrict__ mask, // [B,S,S] f32
    _Float16* __restrict__ outH) {     // [B*S, D] f16 (heads merged)
  __shared__ alignas(32) _Float16 sK[64 * 64];     // [kv][d]
  __shared__ alignas(32) _Float16 sVt[64 * 64];    // [d][kv]
  __shared__ alignas(32) _Float16 sP[8 * 16 * 64]; // per-wave P staging

  const int tid  = threadIdx.x;
  const int lane = tid & 31;
  const int wave = tid >> 5;  // 0..7
  const int qb   = blockIdx.x * 128;
  const int bh   = blockIdx.y;
  const int b    = bh / H_;
  const int h    = bh % H_;
  const int mlo  = lane & 15;
  const int khf  = lane >> 4;

  // Q fragments: two 16x32 A-frags (d 0..31, 32..63), loaded once
  v16h qf[2];
#pragma unroll
  for (int f = 0; f < 2; ++f) {
#pragma unroll
    for (int i = 0; i < 8; ++i) {
      int kb = 16 * khf + 2 * i + f * 32;
      const _Float16* src =
          &qkv[(size_t)(b * S_ + qb + wave * 16 + mlo) * (3 * D_) + h * HD_ + kb];
      qf[f][2 * i]     = src[0];
      qf[f][2 * i + 1] = src[1];
    }
  }

  v8f o[4];
#pragma unroll
  for (int nt = 0; nt < 4; ++nt) o[nt] = zero8();
  float mrow[8], lrow[8];
#pragma unroll
  for (int i = 0; i < 8; ++i) { mrow[i] = -1e30f; lrow[i] = 0.f; }

  const float sc = 0.125f;  // 1/sqrt(64)

  for (int jb = 0; jb < S_; jb += 64) {
    __syncthreads();
    // Stage K tile [kv][d] and V tile [d][kv], both coalesced b128
#pragma unroll
    for (int rep = 0; rep < 2; ++rep) {
      int c = tid + rep * 256;
      int r = c >> 3, q = (c & 7) * 8;
      *(uint4*)&sK[r * 64 + q] =
          *(const uint4*)&qkv[(size_t)(b * S_ + jb + r) * (3 * D_) + D_ + h * HD_ + q];
      *(uint4*)&sVt[r * 64 + q] =
          *(const uint4*)&vT[((size_t)bh * HD_ + r) * S_ + jb + q];
    }
    __syncthreads();

    // S = Q * K^T : B-frag = contiguous 32B read of sK
    v8f s[4];
#pragma unroll
    for (int nt = 0; nt < 4; ++nt) {
      const _Float16* kp = &sK[(nt * 16 + mlo) * 64 + 16 * khf];
      v16h b0 = *(const v16h*)kp;         // d = 0..31
      v16h b1 = *(const v16h*)(kp + 32);  // d = 32..63
      v8f t = __builtin_amdgcn_wmma_f32_16x16x32_f16(
          false, qf[0], false, b0, (short)0, zero8(), false, false);
      s[nt] = __builtin_amdgcn_wmma_f32_16x16x32_f16(
          false, qf[1], false, b1, (short)0, t, false, false);
    }

    // scale + additive mask
#pragma unroll
    for (int nt = 0; nt < 4; ++nt) {
      int kc = jb + nt * 16 + mlo;
#pragma unroll
      for (int i = 0; i < 8; ++i) {
        int qr = qb + wave * 16 + i + 8 * khf;
        float mk = mask[((size_t)b * S_ + qr) * S_ + kc];
        s[nt][i] = s[nt][i] * sc - 1e30f * (1.0f - mk);
      }
    }

    // row max over 4 n-tiles and the 16 lanes holding each row
    float rmax[8];
#pragma unroll
    for (int i = 0; i < 8; ++i)
      rmax[i] = fmaxf(fmaxf(s[0][i], s[1][i]), fmaxf(s[2][i], s[3][i]));
#pragma unroll
    for (int d = 1; d < 16; d <<= 1)
#pragma unroll
      for (int i = 0; i < 8; ++i)
        rmax[i] = fmaxf(rmax[i], __shfl_xor(rmax[i], d, 32));

    // online softmax update
    float corr[8], rsum[8];
#pragma unroll
    for (int i = 0; i < 8; ++i) {
      float mn = fmaxf(mrow[i], rmax[i]);
      corr[i] = __expf(mrow[i] - mn);
      mrow[i] = mn;
      rsum[i] = 0.f;
    }
#pragma unroll
    for (int nt = 0; nt < 4; ++nt)
#pragma unroll
      for (int i = 0; i < 8; ++i) {
        float p = __expf(s[nt][i] - mrow[i]);
        rsum[i] += p;
        sP[wave * 1024 + (i + 8 * khf) * 64 + nt * 16 + mlo] = (_Float16)p;
      }
#pragma unroll
    for (int d = 1; d < 16; d <<= 1)
#pragma unroll
      for (int i = 0; i < 8; ++i) rsum[i] += __shfl_xor(rsum[i], d, 32);
#pragma unroll
    for (int i = 0; i < 8; ++i) lrow[i] = lrow[i] * corr[i] + rsum[i];
#pragma unroll
    for (int nt = 0; nt < 4; ++nt)
#pragma unroll
      for (int i = 0; i < 8; ++i) o[nt][i] *= corr[i];

    // reload P as A-fragments (wave-private region; DScnt keeps order)
    v16h pf[2];
#pragma unroll
    for (int f = 0; f < 2; ++f)
#pragma unroll
      for (int i = 0; i < 8; ++i) {
        int kb = 16 * khf + 2 * i + f * 32;
        pf[f][2 * i]     = sP[wave * 1024 + mlo * 64 + kb];
        pf[f][2 * i + 1] = sP[wave * 1024 + mlo * 64 + kb + 1];
      }

    // O += P * V : B-frag = contiguous 32B read of sVt [d][kv]
#pragma unroll
    for (int nt = 0; nt < 4; ++nt) {
      const _Float16* vp = &sVt[(nt * 16 + mlo) * 64 + 16 * khf];
      v16h bv0 = *(const v16h*)vp;         // kv = 0..31
      v16h bv1 = *(const v16h*)(vp + 32);  // kv = 32..63
      o[nt] = __builtin_amdgcn_wmma_f32_16x16x32_f16(
          false, pf[0], false, bv0, (short)0, o[nt], false, false);
      o[nt] = __builtin_amdgcn_wmma_f32_16x16x32_f16(
          false, pf[1], false, bv1, (short)0, o[nt], false, false);
    }
  }

  // normalize and write merged-head output (f16 for projection GEMM)
#pragma unroll
  for (int nt = 0; nt < 4; ++nt) {
    int col = h * HD_ + nt * 16 + mlo;
#pragma unroll
    for (int i = 0; i < 8; ++i) {
      int qr = qb + wave * 16 + i + 8 * khf;
      outH[((size_t)(b * S_ + qr)) * D_ + col] = (_Float16)(o[nt][i] / lrow[i]);
    }
  }
}

// ---------------------------------------------------------------------------
extern "C" void kernel_launch(void* const* d_in, const int* in_sizes, int n_in,
                              void* d_out, int out_size, void* d_ws, size_t ws_size,
                              hipStream_t stream) {
  const float* hs   = (const float*)d_in[0];  // [B,S,D]
  const float* mask = (const float*)d_in[1];  // [B,S,S]
  const float* cw   = (const float*)d_in[2];  // [D,3D]
  const float* cb   = (const float*)d_in[3];  // [3D]
  const float* pw   = (const float*)d_in[4];  // [D,D]
  const float* pb   = (const float*)d_in[5];  // [D]
  float* out = (float*)d_out;

  const size_t MB = (size_t)B_ * S_;  // 8192 rows
  _Float16* h16    = (_Float16*)d_ws;                 // [8192,1024]
  _Float16* wA16t  = h16    + MB * D_;                // [3072,1024] (N,K)
  _Float16* wP16t  = wA16t  + (size_t)3 * D_ * D_;    // [1024,1024] (N,K)
  _Float16* qkv16  = wP16t  + (size_t)D_ * D_;        // [8192,3072]
  _Float16* att16  = qkv16  + MB * 3 * D_;            // [8192,1024]
  _Float16* vT16   = att16  + MB * D_;                // [B*H*HD, S]

  // 1) activations f32 -> f16 ; weights f32 -> f16 transposed
  {
    int n4 = (int)(MB * D_ / 4);
    cvt_f32_to_f16<<<(n4 + 255) / 256, 256, 0, stream>>>(hs, h16, n4);
    transpose_cvt_f32_f16<<<dim3(3 * D_ / 32, D_ / 32), 256, 0, stream>>>(
        cw, wA16t, D_, 3 * D_);
    transpose_cvt_f32_f16<<<dim3(D_ / 32, D_ / 32), 256, 0, stream>>>(
        pw, wP16t, D_, D_);
  }

  // 2) QKV projection -> f16 qkv
  gemm_f16_wmma<<<dim3(3 * D_ / 64, MB / 256), 256, 0, stream>>>(
      h16, wA16t, cb, qkv16, nullptr, (int)MB, 3 * D_, D_, 1);

  // 3) V transpose for contiguous P*V fragments
  transpose_v_f16<<<dim3(S_ / 64, B_ * H_), 256, 0, stream>>>(qkv16, vT16);

  // 4) fused flash attention -> f16 merged-head output
  flash_attn_wmma<<<dim3(S_ / 128, B_ * H_), 256, 0, stream>>>(
      qkv16, vT16, mask, att16);

  // 5) output projection -> f32 d_out
  gemm_f16_wmma<<<dim3(D_ / 64, MB / 256), 256, 0, stream>>>(
      att16, wP16t, pb, nullptr, out, (int)MB, D_, D_, 0);
}
